// GINModel_12463995093414
// MI455X (gfx1250) — matrix operand order repaired and checked by reading, not compiled, not run on hardware
//
#include <hip/hip_runtime.h>
#include <hip/hip_bf16.h>

// ---------------------------------------------------------------------------
// GIN model on MI455X (gfx1250, wave32).
//  * Edge BatchNorm folded analytically (needs only mean(3) + 3x3 second
//    moments of edge_attr) -> per-edge MLP relu(t@W1'+b1') @ W2 + b2 is fused
//    with the segment_sum scatter; the [1.2M x 64] edge tensor never exists.
//  * All N=64 GEMMs use v_wmma_f32_16x16x32_f16 (f32 accumulate), A staged
//    to LDS as f16 row-major, read as contiguous k = 16h + s fragments, B = transposed f16
//    weights resident in L0/L2.  GEMM templated on padded-K: fragments live
//    in VGPRs (verified: ds_load_b128 -> v_wmma, no scratch).
//  * Bounds guards are wave-uniform branches (full-tile fast path) so the
//    hot scatter/store loops are free of per-lane EXEC-mask manipulation.
//  * Node feature buffers (12.8MB) are L2-resident (192MB L2): both
//    segment-sum scatters run as on-chip f32 atomics.
// ---------------------------------------------------------------------------

typedef __attribute__((ext_vector_type(16))) _Float16 v16h;
typedef __attribute__((ext_vector_type(8)))  _Float16 v8h;
typedef __attribute__((ext_vector_type(8)))  float    v8f;

#define GN_N      50000
#define GN_E      1200000
#define GN_INDIM  140
#define GN_KPIN   160      // IN_DIM padded to multiple of 32
#define GN_EMB    64
#define GN_L      4
#define GN_EPS    1e-5f

// ---- WMMA helpers ----------------------------------------------------------

__device__ __forceinline__ v8f wmma_f16(v16h a, v16h b, v8f c) {
  // D = A(16x32 f16) * B(32x16 f16) + C(16x16 f32)
  return __builtin_amdgcn_wmma_f32_16x16x32_f16(false, a, false, b,
                                                (short)0, c, false, false);
}

// A fragment (16x32 f16 tile, K-offset k0) from LDS row-major (rowStride halves).
// Layout: lane<16 holds row=lane, K = k0+[0..15];
// lane>=16 holds row=lane-16, K = k0+[16..31] (matches B fragment order).
__device__ __forceinline__ v16h lds_a_frag(const _Float16* base, int rowStride,
                                           int k0, int lane) {
  int row = lane & 15;
  int kb  = k0 + ((lane & 16) ? 16 : 0);
  const _Float16* p = base + row * rowStride + kb;
  v8h lo = *(const v8h*)p;
  v8h hi = *(const v8h*)(p + 8);
  return __builtin_shufflevector(lo, hi, 0,1,2,3,4,5,6,7,8,9,10,11,12,13,14,15);
}

// B fragment (32x16 f16, columns ntile*16..+15, K-offset k0) from Bt stored
// transposed row-major [N][KP]. Layout: lane<16 holds K=k0..k0+15 of col n;
// lane>=16 holds K=k0+16..k0+31 of col n-16.  Contiguous 32B per lane.
__device__ __forceinline__ v16h glb_b_frag(const _Float16* bt, int KP,
                                           int ntile, int k0, int lane) {
  int col = ntile * 16 + (lane & 15);
  int kb  = k0 + ((lane & 16) ? 16 : 0);
  return *(const v16h*)(bt + (size_t)col * KP + kb);
}

// ---- small utility kernels -------------------------------------------------

__global__ void k_zero_f32(float* p, int n) {
  int i = blockIdx.x * blockDim.x + threadIdx.x;
  if (i < n) p[i] = 0.f;
}

__global__ void k_copy_f4(const float* __restrict__ src, float* __restrict__ dst,
                          long n4) {
  long i = (long)blockIdx.x * blockDim.x + threadIdx.x;
  long stride = (long)gridDim.x * blockDim.x;
  for (; i < n4; i += stride)
    ((float4*)dst)[i] = ((const float4*)src)[i];
}

// Convert all weight matrices to f16, transposed to [N][K] for the B layout.
__global__ void k_prep_weights(const float* __restrict__ ee_w2,   // [L][64][64]
                               const float* __restrict__ mlp_w1,  // [L][64][64]
                               const float* __restrict__ mlp_w2,  // [L][64][64]
                               const float* __restrict__ lin_w,   // [140][64]
                               _Float16* __restrict__ w2t_ee,     // [L][64][64]
                               _Float16* __restrict__ w1t_m,      // [L][64][64]
                               _Float16* __restrict__ w2t_m,      // [L][64][64]
                               _Float16* __restrict__ lin_wt) {   // [64][160]
  const int per = GN_L * GN_EMB * GN_EMB;  // 16384
  for (int i = threadIdx.x; i < per; i += blockDim.x) {
    int l = i >> 12, r = i & 4095;
    int n = r >> 6, k = r & 63;   // output [l][n][k] = in [l][k][n]
    w2t_ee[i] = (_Float16)ee_w2 [(size_t)l*4096 + k*64 + n];
    w1t_m [i] = (_Float16)mlp_w1[(size_t)l*4096 + k*64 + n];
    w2t_m [i] = (_Float16)mlp_w2[(size_t)l*4096 + k*64 + n];
  }
  for (int i = threadIdx.x; i < GN_EMB * GN_KPIN; i += blockDim.x) {
    int n = i / GN_KPIN, k = i % GN_KPIN;
    lin_wt[i] = (k < GN_INDIM) ? (_Float16)lin_w[(size_t)k*GN_EMB + n]
                               : (_Float16)0.0f;
  }
}

// mean(3) + second moments(6) of edge_attr -> estats[0..8]
__global__ void k_edge_stats(const float* __restrict__ ea, float* estats, int E) {
  float s0=0,s1=0,s2=0,s3=0,s4=0,s5=0,s6=0,s7=0,s8=0;
  long i = (long)blockIdx.x * blockDim.x + threadIdx.x;
  long stride = (long)gridDim.x * blockDim.x;
  for (; i < E; i += stride) {
    float a = ea[i*3+0], b = ea[i*3+1], c = ea[i*3+2];
    s0+=a; s1+=b; s2+=c;
    s3+=a*a; s4+=a*b; s5+=a*c; s6+=b*b; s7+=b*c; s8+=c*c;
  }
  #pragma unroll
  for (int m = 16; m >= 1; m >>= 1) {
    s0 += __shfl_xor(s0,m,32); s1 += __shfl_xor(s1,m,32); s2 += __shfl_xor(s2,m,32);
    s3 += __shfl_xor(s3,m,32); s4 += __shfl_xor(s4,m,32); s5 += __shfl_xor(s5,m,32);
    s6 += __shfl_xor(s6,m,32); s7 += __shfl_xor(s7,m,32); s8 += __shfl_xor(s8,m,32);
  }
  if ((threadIdx.x & 31) == 0) {
    atomicAdd(&estats[0],s0); atomicAdd(&estats[1],s1); atomicAdd(&estats[2],s2);
    atomicAdd(&estats[3],s3); atomicAdd(&estats[4],s4); atomicAdd(&estats[5],s5);
    atomicAdd(&estats[6],s6); atomicAdd(&estats[7],s7); atomicAdd(&estats[8],s8);
  }
}

// Fold BN into the 3->64 edge linear for all layers: u = relu(t@W1' + b1')
__global__ void k_fold_edge(const float* __restrict__ estats,
                            const float* __restrict__ ee_w1,  // [L][3][64]
                            const float* __restrict__ ee_b1,  // [L][64]
                            const float* __restrict__ ee_g1,
                            const float* __restrict__ ee_be1,
                            float* __restrict__ w1f,          // [L][3][64]
                            float* __restrict__ b1f,          // [L][64]
                            int E) {
  int t = threadIdx.x;                      // 256 = L*64
  int l = t >> 6, j = t & 63;
  float inv = 1.f / (float)E;
  float m0 = estats[0]*inv, m1 = estats[1]*inv, m2 = estats[2]*inv;
  float cxx = estats[3]*inv - m0*m0, cxy = estats[4]*inv - m0*m1;
  float cxz = estats[5]*inv - m0*m2, cyy = estats[6]*inv - m1*m1;
  float cyz = estats[7]*inv - m1*m2, czz = estats[8]*inv - m2*m2;
  const float* w = ee_w1 + (size_t)l*3*64;
  float w0 = w[j], w1 = w[64+j], w2 = w[128+j];
  float b  = ee_b1[l*64+j];
  float mu  = m0*w0 + m1*w1 + m2*w2 + b;    // mean of pre-activation
  float var = w0*w0*cxx + w1*w1*cyy + w2*w2*czz
            + 2.f*(w0*w1*cxy + w0*w2*cxz + w1*w2*cyz);
  float s = ee_g1[l*64+j] * rsqrtf(var + GN_EPS);
  w1f[(l*3+0)*64+j] = w0 * s;
  w1f[(l*3+1)*64+j] = w1 * s;
  w1f[(l*3+2)*64+j] = w2 * s;
  b1f[l*64+j] = (b - mu) * s + ee_be1[l*64+j];
}

// ---- generic WMMA GEMM: out[M x 64] = A[M x K](f32) @ Bt'[64 x KP](f16) + bias
// Templated on padded K so all fragment arrays stay in VGPRs (no scratch).
template <int KP>
__global__ __launch_bounds__(128)
void k_gemm_n64(const float* __restrict__ A, const _Float16* __restrict__ Bt,
                const float* __restrict__ bias, float* __restrict__ out,
                int M, int K) {
  constexpr int STRIDE = KP + 16;           // halves; keeps 16B row alignment
  constexpr int KSTEPS = KP / 32;
  constexpr int HALFK  = KP / 2;
  __shared__ _Float16 sA[4][16 * STRIDE];
  int wave = threadIdx.x >> 5;
  int lane = threadIdx.x & 31;
  int mbase = (blockIdx.x * 4 + wave) * 16;
  // stage 16 x KP A-tile to LDS as f16 (zero-pad K..KP and rows >= M)
  {
    int row  = lane & 15;
    int half = lane >> 4;
    int grow = mbase + row;
    bool rv  = grow < M;
    const float* ap = A + (size_t)grow * K;
    int k0 = half * HALFK;
    #pragma unroll
    for (int kk = 0; kk < HALFK; kk += 8) {
      v8h pack;
      #pragma unroll
      for (int q = 0; q < 2; ++q) {
        int kq = k0 + kk + q * 4;
        if (rv && kq + 4 <= K) {            // 16B-aligned full float4
          float4 v = *(const float4*)(ap + kq);
          pack[q*4+0] = (_Float16)v.x; pack[q*4+1] = (_Float16)v.y;
          pack[q*4+2] = (_Float16)v.z; pack[q*4+3] = (_Float16)v.w;
        } else {
          #pragma unroll
          for (int j = 0; j < 4; ++j) {
            float v = (rv && kq + j < K) ? ap[kq + j] : 0.f;
            pack[q*4+j] = (_Float16)v;
          }
        }
      }
      *(v8h*)&sA[wave][row * STRIDE + k0 + kk] = pack;
    }
  }
  __syncthreads();
  const _Float16* sa = sA[wave];
  v16h afrag[KSTEPS];
  #pragma unroll
  for (int kt = 0; kt < KSTEPS; ++kt)
    afrag[kt] = lds_a_frag(sa, STRIDE, kt * 32, lane);
  int rbase = (lane & 16) ? 8 : 0;
  bool fullTile = (mbase + 16 <= M);        // wave-uniform
  #pragma unroll
  for (int nt = 0; nt < 4; ++nt) {
    int col = nt * 16 + (lane & 15);
    float bv = bias[col];
    v8f acc;
    #pragma unroll
    for (int r = 0; r < 8; ++r) acc[r] = bv;
    #pragma unroll
    for (int kt = 0; kt < KSTEPS; ++kt)
      acc = wmma_f16(afrag[kt], glb_b_frag(Bt, KP, nt, kt * 32, lane), acc);
    float* op = out + (size_t)(mbase + rbase) * 64 + col;
    if (fullTile) {                         // branch-free hot path
      #pragma unroll
      for (int r = 0; r < 8; ++r) op[r * 64] = acc[r];
    } else {
      #pragma unroll
      for (int r = 0; r < 8; ++r)
        if (mbase + rbase + r < M) op[r * 64] = acc[r];
    }
  }
}

// ---- fused edge encoder + segment_sum scatter ------------------------------
// Per 16-edge tile: u = relu(t@W1'+b1') (VALU, ->LDS f16), e = u@W2+b2 (WMMA),
// atomicAdd rows into hh[dst].
__global__ __launch_bounds__(256)
void k_edge_encode_scatter(const float* __restrict__ edge_attr,
                           const int* __restrict__ dst,
                           const float* __restrict__ w1f,   // [3][64]
                           const float* __restrict__ b1f,   // [64]
                           const _Float16* __restrict__ w2t,// [64][64]
                           const float* __restrict__ b2,    // [64]
                           float* __restrict__ hh, int E) {
  __shared__ _Float16 sU[8][16 * 80];
  int wave = threadIdx.x >> 5;
  int lane = threadIdx.x & 31;
  long tile  = (long)blockIdx.x * 8 + wave;
  long ebase = tile * 16;
  bool fullTile = (ebase + 16 <= (long)E);  // wave-uniform
  // ---- hidden layer u: lane (row = lane&15, feature-chunk = lane>>4) ----
  int erow  = lane & 15;
  int chunk = lane >> 4;
  long eid  = ebase + erow;
  float t0 = 0.f, t1 = 0.f, t2 = 0.f;
  if (eid < E) {
    t0 = edge_attr[eid*3+0]; t1 = edge_attr[eid*3+1]; t2 = edge_attr[eid*3+2];
  }
  _Float16* urow = &sU[wave][erow * 80 + chunk * 32];
  #pragma unroll
  for (int j0 = 0; j0 < 32; j0 += 8) {
    v8h pack;
    #pragma unroll
    for (int jj = 0; jj < 8; ++jj) {
      int j = chunk * 32 + j0 + jj;
      float u = fmaf(t2, w1f[128+j], fmaf(t1, w1f[64+j], fmaf(t0, w1f[j], b1f[j])));
      pack[jj] = (_Float16)(u > 0.f ? u : 0.f);
    }
    *(v8h*)(urow + j0) = pack;
  }
  __syncthreads();
  // ---- 16x64 = (16x64) @ (64x64) via 8 WMMAs ----
  int rbase = (lane & 16) ? 8 : 0;
  v16h a0 = lds_a_frag(sU[wave], 80, 0,  lane);
  v16h a1 = lds_a_frag(sU[wave], 80, 32, lane);
  v8f accs[4];
  #pragma unroll
  for (int nt = 0; nt < 4; ++nt) {
    int col = nt * 16 + (lane & 15);
    float bv = b2[col];
    v8f acc;
    #pragma unroll
    for (int r = 0; r < 8; ++r) acc[r] = bv;
    acc = wmma_f16(a0, glb_b_frag(w2t, 64, nt, 0,  lane), acc);
    acc = wmma_f16(a1, glb_b_frag(w2t, 64, nt, 32, lane), acc);
    accs[nt] = acc;
  }
  // ---- scatter: wave-uniform branch keeps the hot path EXEC-clean ----
  int colb = lane & 15;
  if (fullTile) {
    int drow[8];
    #pragma unroll
    for (int r = 0; r < 8; ++r) drow[r] = dst[ebase + rbase + r];
    #pragma unroll
    for (int nt = 0; nt < 4; ++nt) {
      #pragma unroll
      for (int r = 0; r < 8; ++r)
        atomicAdd(&hh[(size_t)drow[r] * 64 + nt * 16 + colb], accs[nt][r]);
    }
  } else {
    #pragma unroll
    for (int r = 0; r < 8; ++r) {
      long e = ebase + rbase + r;
      if (e < E) {
        int d = dst[e];
        #pragma unroll
        for (int nt = 0; nt < 4; ++nt)
          atomicAdd(&hh[(size_t)d * 64 + nt * 16 + colb], accs[nt][r]);
      }
    }
  }
}

// z[dst] += hh[src]  (half-wave per edge, float4 chunks; all L2-resident)
__global__ void k_msg_scatter(const float* __restrict__ hh, float* __restrict__ z,
                              const int* __restrict__ src,
                              const int* __restrict__ dst, int E) {
  long total = (long)E * 16;
  long i = (long)blockIdx.x * blockDim.x + threadIdx.x;
  long stride = (long)gridDim.x * blockDim.x;
  for (; i < total; i += stride) {
    long e  = i >> 4;
    int  c4 = (int)(i & 15) * 4;
    int  s  = src[e], d = dst[e];
    float4 v = *(const float4*)(hh + (size_t)s * 64 + c4);
    float* zp = z + (size_t)d * 64 + c4;
    atomicAdd(zp + 0, v.x); atomicAdd(zp + 1, v.y);
    atomicAdd(zp + 2, v.z); atomicAdd(zp + 3, v.w);
  }
}

// per-column sum & sumsq of y[M x 64] -> cs[0..63]=sum, cs[64..127]=sumsq
__global__ void k_colstats(const float* __restrict__ y, float* cs, int M) {
  int col  = threadIdx.x & 63;
  int rgrp = threadIdx.x >> 6;                // 0..3
  float s = 0.f, s2 = 0.f;
  for (int r = blockIdx.x * 4 + rgrp; r < M; r += gridDim.x * 4) {
    float v = y[(size_t)r * 64 + col];
    s += v; s2 += v * v;
  }
  __shared__ float red[2][4][64];
  red[0][rgrp][col] = s; red[1][rgrp][col] = s2;
  __syncthreads();
  if (rgrp == 0) {
    s  = red[0][0][col] + red[0][1][col] + red[0][2][col] + red[0][3][col];
    s2 = red[1][0][col] + red[1][1][col] + red[1][2][col] + red[1][3][col];
    atomicAdd(&cs[col], s);
    atomicAdd(&cs[64 + col], s2);
  }
}

// v = relu(bn(y)); resid ? resid += v : y = v
__global__ void k_bn_relu(float* __restrict__ y, const float* __restrict__ cs,
                          const float* __restrict__ g, const float* __restrict__ be,
                          float* __restrict__ resid, int M) {
  long n = (long)M * 64;
  long i = (long)blockIdx.x * blockDim.x + threadIdx.x;
  long stride = (long)gridDim.x * blockDim.x;
  float invM = 1.f / (float)M;
  for (; i < n; i += stride) {
    int col = (int)(i & 63);
    float mu  = cs[col] * invM;
    float var = cs[64 + col] * invM - mu * mu;
    float v = (y[i] - mu) * rsqrtf(var + GN_EPS) * g[col] + be[col];
    v = v > 0.f ? v : 0.f;
    if (resid) resid[i] += v; else y[i] = v;
  }
}

// out[i] = h[i] . w + b
__global__ void k_pred(const float* __restrict__ h, const float* __restrict__ w,
                       const float* __restrict__ b, float* __restrict__ out, int M) {
  int n = blockIdx.x * blockDim.x + threadIdx.x;
  if (n < M) {
    float s = b[0];
    const float* hp = h + (size_t)n * 64;
    #pragma unroll
    for (int j = 0; j < 64; ++j) s = fmaf(hp[j], w[j], s);
    out[n] = s;
  }
}

// ---------------------------------------------------------------------------

extern "C" void kernel_launch(void* const* d_in, const int* in_sizes, int n_in,
                              void* d_out, int out_size, void* d_ws, size_t ws_size,
                              hipStream_t stream) {
  (void)in_sizes; (void)n_in; (void)out_size; (void)ws_size;
  const float* x      = (const float*)d_in[0];
  const int*   eidx   = (const int*)  d_in[1];
  const float* eattr  = (const float*)d_in[2];
  const float* lin_w  = (const float*)d_in[3];
  const float* lin_b  = (const float*)d_in[4];
  const float* ee_w1  = (const float*)d_in[5];
  const float* ee_b1  = (const float*)d_in[6];
  const float* ee_g1  = (const float*)d_in[7];
  const float* ee_be1 = (const float*)d_in[8];
  const float* ee_w2  = (const float*)d_in[9];
  const float* ee_b2  = (const float*)d_in[10];
  const float* mw1    = (const float*)d_in[11];
  const float* mb1    = (const float*)d_in[12];
  const float* mg1    = (const float*)d_in[13];
  const float* mbe1   = (const float*)d_in[14];
  const float* mw2    = (const float*)d_in[15];
  const float* mb2    = (const float*)d_in[16];
  const float* mg2    = (const float*)d_in[17];
  const float* mbe2   = (const float*)d_in[18];
  const float* pw     = (const float*)d_in[19];
  const float* pb     = (const float*)d_in[20];
  float* out = (float*)d_out;

  const int N = GN_N, E = GN_E;
  const int* src = eidx;
  const int* dstp = eidx + E;

  // workspace carve-out (256B aligned)
  char* ws = (char*)d_ws; size_t off = 0;
  auto carve = [&](size_t bytes) {
    void* p = ws + off; off = (off + bytes + 255) & ~(size_t)255; return p;
  };
  float*    h      = (float*)   carve((size_t)N * 64 * 4);
  float*    hh     = (float*)   carve((size_t)N * 64 * 4);
  float*    y      = (float*)   carve((size_t)N * 64 * 4);
  float*    y2     = (float*)   carve((size_t)N * 64 * 4);
  _Float16* w2t_ee = (_Float16*)carve((size_t)GN_L * 64 * 64 * 2);
  _Float16* w1t_m  = (_Float16*)carve((size_t)GN_L * 64 * 64 * 2);
  _Float16* w2t_m  = (_Float16*)carve((size_t)GN_L * 64 * 64 * 2);
  _Float16* lin_wt = (_Float16*)carve((size_t)64 * GN_KPIN * 2);
  float*    w1f    = (float*)   carve((size_t)GN_L * 3 * 64 * 4);
  float*    b1f    = (float*)   carve((size_t)GN_L * 64 * 4);
  float*    estats = (float*)   carve(16 * 4);
  float*    cstats = (float*)   carve(128 * 4);

  const long n4 = (long)N * 64 / 4;
  const int gemmGrid = (N + 63) / 64;
  const int edgeTiles = (E + 15) / 16;
  const int edgeGrid  = (edgeTiles + 7) / 8;

  // prep: f16 weights, edge stats, BN fold, input projection
  k_zero_f32<<<1, 64, 0, stream>>>(estats, 16);
  k_prep_weights<<<1, 256, 0, stream>>>(ee_w2, mw1, mw2, lin_w,
                                        w2t_ee, w1t_m, w2t_m, lin_wt);
  k_edge_stats<<<128, 256, 0, stream>>>(eattr, estats, E);
  k_fold_edge<<<1, 256, 0, stream>>>(estats, ee_w1, ee_b1, ee_g1, ee_be1,
                                     w1f, b1f, E);
  k_gemm_n64<GN_KPIN><<<gemmGrid, 128, 0, stream>>>(x, lin_wt, lin_b, h,
                                                    N, GN_INDIM);

  for (int l = 0; l < GN_L; ++l) {
    // hh = h + segment_sum(edge_mlp(edge_attr), dst)
    k_copy_f4<<<1024, 256, 0, stream>>>(h, hh, n4);
    k_edge_encode_scatter<<<edgeGrid, 256, 0, stream>>>(
        eattr, dstp, w1f + (size_t)l * 192, b1f + (size_t)l * 64,
        w2t_ee + (size_t)l * 4096, ee_b2 + (size_t)l * 64, hh, E);
    // y = hh + segment_sum(hh[src], dst)
    k_copy_f4<<<1024, 256, 0, stream>>>(hh, y, n4);
    k_msg_scatter<<<4096, 256, 0, stream>>>(hh, y, src, dstp, E);
    // y2 = relu(bn(y @ mlp_w1 + b1))
    k_gemm_n64<64><<<gemmGrid, 128, 0, stream>>>(y, w1t_m + (size_t)l * 4096,
                                                 mb1 + l * 64, y2, N, 64);
    k_zero_f32<<<1, 128, 0, stream>>>(cstats, 128);
    k_colstats<<<256, 256, 0, stream>>>(y2, cstats, N);
    k_bn_relu<<<2048, 256, 0, stream>>>(y2, cstats, mg1 + l * 64, mbe1 + l * 64,
                                        nullptr, N);
    // h += relu(bn(y2 @ mlp_w2 + b2))
    k_gemm_n64<64><<<gemmGrid, 128, 0, stream>>>(y2, w2t_m + (size_t)l * 4096,
                                                 mb2 + l * 64, y, N, 64);
    k_zero_f32<<<1, 128, 0, stream>>>(cstats, 128);
    k_colstats<<<256, 256, 0, stream>>>(y, cstats, N);
    k_bn_relu<<<2048, 256, 0, stream>>>(y, cstats, mg2 + l * 64, mbe2 + l * 64,
                                        h, N);
  }

  k_pred<<<(N + 255) / 256, 256, 0, stream>>>(h, pw, pb, out, N);
}
